// HierarchicalBackmappingModule_84490596647421
// MI455X (gfx1250) — compile-verified, matrix operand order checked by reading, not run on hardware
//
#include <hip/hip_runtime.h>
#include <cmath>

typedef __attribute__((ext_vector_type(16))) __bf16 v16b;
typedef __attribute__((ext_vector_type(8)))  float  v8f;

#define WAVES 4
#define TPB   128

// packed-weight dword offsets in workspace
#define OFF_WL0A  0
#define OFF_WL0B  2048
#define OFF_WL0C  10240
#define OFF_WL1A  18432
#define OFF_WL1B  30720
#define OFF_WL1C  38912
#define OFF_WENV0 47104
#define OFF_WENV1 55296
#define OFF_WLS0  59392
#define OFF_WLS1  60416
#define OFF_WLV0  61440
#define OFF_WLV1  62464
#define WS_FLOAT_BASE 65536

#define DSWAIT() asm volatile("s_wait_dscnt 0" ::: "memory")
#define LDWAIT() asm volatile("s_wait_loadcnt 0" ::: "memory")

__device__ __forceinline__ unsigned f2bf(float f) {
  unsigned x = __float_as_uint(f);
  return (x + 0x7fffu + ((x >> 16) & 1u)) >> 16;  // RNE f32 -> bf16
}
__device__ __forceinline__ unsigned pk2(float a, float b) {
  return f2bf(a) | (f2bf(b) << 16);
}
__device__ __forceinline__ float silu(float x) { return x / (1.f + __expf(-x)); }

union FragU { unsigned u[8]; v16b v; };

// A fragment: 16x32 bf16, lane holds row m=lane%16, h=lane/16:
//   vgpr j<4 : K = 8h+2j,+1   ; vgpr j>=4 : K = 16+8h+2(j-4),+1
// act buffer stores bf16 pairs row-major: dword t covers K=2t,2t+1
__device__ __forceinline__ v16b ldA(const unsigned* act, int strideDw, int lane, int kt) {
  int m = lane & 15, h = lane >> 4;
  const unsigned* p = act + m * strideDw + kt * 16 + 4 * h;
  FragU f;
#pragma unroll
  for (int j = 0; j < 4; ++j) { f.u[j] = p[j]; f.u[4 + j] = p[8 + j]; }
  return f.v;
}
// B fragment: 32x16 bf16, lane holds col n=lane%16, K = 8h+2j,+1 (j<4), 16+8h+2(j-4),+1 (j>=4), h=lane/16
// pre-packed in ws: dword index = ((nt*KT+kt)*32+lane)*8 + j  -> 2x b128 per lane
__device__ __forceinline__ v16b ldB(const unsigned* w, int KT, int kt, int nt, int lane) {
  const unsigned* p = w + ((size_t)((nt * KT) + kt) * 32 + lane) * 8;
  FragU f;
#pragma unroll
  for (int j = 0; j < 8; ++j) f.u[j] = p[j];
  return f.v;
}

template <int KT, int NT>
__device__ __forceinline__ void gemm(const unsigned* act, int strideDw, const unsigned* w,
                                     int lane, v8f* C) {
#pragma unroll
  for (int nt = 0; nt < NT; ++nt)
#pragma unroll
    for (int r = 0; r < 8; ++r) C[nt][r] = 0.f;
#pragma unroll
  for (int kt = 0; kt < KT; ++kt) {
    v16b a = ldA(act, strideDw, lane, kt);
#pragma unroll
    for (int nt = 0; nt < NT; ++nt) {
      v16b b = ldB(w, KT, kt, nt, lane);
      C[nt] = __builtin_amdgcn_wmma_f32_16x16x32_bf16(false, a, false, b, (short)0,
                                                      C[nt], false, false);
    }
  }
}

// C/D layout: lane holds col n=nt*16+lane%16, rows m = 8*(lane/16)+r
__device__ __forceinline__ void stActRow(unsigned short* acth, int strideH, int lane, int nt,
                                         const v8f& c) {
  int n = nt * 16 + (lane & 15), h = lane >> 4;
#pragma unroll
  for (int r = 0; r < 8; ++r) acth[(8 * h + r) * strideH + n] = (unsigned short)f2bf(c[r]);
}

__device__ __forceinline__ float cutoff_of(float r) {
  float u = r * (1.0f / 6.0f);
  float u2 = u * u, u4 = u2 * u2, u6 = u4 * u2;
  float f = 1.0f - 28.0f * u6 + 48.0f * u6 * u - 21.0f * u6 * u2;
  return (u < 1.0f) ? f : 0.0f;
}

// Build out_s (16x64 bf16, into os) and out_v[3] (16x64 bf16 each, into ov),
// reading fs/fv (prev layer) and gathered env_lin[center].
__device__ __forceinline__ void build_out_tiles(int lane, int e0, const int* ctr,
                                                const float* __restrict__ fsBuf,
                                                const float* __restrict__ fvBuf,
                                                const float* __restrict__ envLin,
                                                unsigned* os, int osStride, unsigned* ov) {
  const float inv_s3 = 0.57735026918962576f;
  for (int i = lane; i < 512; i += 32) {
    int m = i >> 5, t = i & 31;
    int e = e0 + m, c = ctr[m];
    float vs[2], v0[2], v1[2], v2[2];
#pragma unroll
    for (int q = 0; q < 2; ++q) {
      int K = 2 * t + q, p = K >> 5, k = K & 31;
      float fsv = fsBuf[(size_t)e * 32 + k];
      float s2 = envLin[(size_t)c * 128 + k];
      float fva = fvBuf[(size_t)e * 96 + k * 3 + 0];
      float fvb = fvBuf[(size_t)e * 96 + k * 3 + 1];
      float fvc = fvBuf[(size_t)e * 96 + k * 3 + 2];
      float v2a = envLin[(size_t)c * 128 + 32 + k * 3 + 0];
      float v2b = envLin[(size_t)c * 128 + 32 + k * 3 + 1];
      float v2c = envLin[(size_t)c * 128 + 32 + k * 3 + 2];
      if (p == 0) {
        vs[q] = fsv * s2;
        v0[q] = fsv * v2a; v1[q] = fsv * v2b; v2[q] = fsv * v2c;
      } else {
        vs[q] = (fva * v2a + fvb * v2b + fvc * v2c) * inv_s3;
        v0[q] = fva * s2; v1[q] = fvb * s2; v2[q] = fvc * s2;
      }
    }
    os[m * osStride + t] = pk2(vs[0], vs[1]);
    ov[0 * 512 + m * 32 + t] = pk2(v0[0], v0[1]);
    ov[1 * 512 + m * 32 + t] = pk2(v1[0], v1[1]);
    ov[2 * 512 + m * 32 + t] = pk2(v2[0], v2[1]);
  }
}

// ---------------- weight packing ----------------
__global__ void k_pack(const float* __restrict__ src, unsigned* __restrict__ dst, int KT,
                       int srcK, int Ncols, float scale) {
  int i = blockIdx.x * blockDim.x + threadIdx.x;
  int total = KT * (Ncols >> 4) * 256;
  if (i >= total) return;
  int j = i & 7, lane = (i >> 3) & 31, q = i >> 8;
  int kt = q % KT, nt = q / KT;
  int h = lane >> 4, n = nt * 16 + (lane & 15);
  int k0 = kt * 32 + (j < 4 ? 8 * h + 2 * j : 16 + 8 * h + 2 * (j - 4));
  float a = (k0 < srcK) ? src[(size_t)k0 * Ncols + n] * scale : 0.f;
  float b = (k0 + 1 < srcK) ? src[(size_t)(k0 + 1) * Ncols + n] * scale : 0.f;
  dst[i] = pk2(a, b);
}

__global__ void k_zero(float* __restrict__ p, int n) {
  int i = blockIdx.x * blockDim.x + threadIdx.x;
  if (i < n) p[i] = 0.f;
}

// ---------------- edge kernel, layer 0 ----------------
__global__ __launch_bounds__(TPB) void k_edge0(
    const float* __restrict__ edge_attrs, const float* __restrict__ edge_embed,
    const float* __restrict__ node_attrs, const float* __restrict__ edge_lengths,
    const int* __restrict__ edge_index, const unsigned* __restrict__ wPack,
    float* __restrict__ out, float* __restrict__ fsBuf, float* __restrict__ fvBuf,
    float* __restrict__ envRaw, int E) {
  int wave = threadIdx.x >> 5, lane = threadIdx.x & 31;
  int e0 = (blockIdx.x * WAVES + wave) * 16;
  if (e0 >= E) return;

  __shared__ unsigned actS[WAVES][16 * 64];
  __shared__ int ctrS[WAVES][16], nbrS[WAVES][16];
  __shared__ float es0S[WAVES][16], ev0S[WAVES][48], cutS[WAVES][16];

  unsigned* act = actS[wave];
  unsigned short* acth = (unsigned short*)act;
  int h = lane >> 4, l16 = lane & 15;

  if (lane < 16) {
    int e = e0 + lane;
    ctrS[wave][lane] = edge_index[e];
    nbrS[wave][lane] = edge_index[E + e];
    es0S[wave][lane] = edge_attrs[(size_t)e * 4 + 0];
    ev0S[wave][lane * 3 + 0] = edge_attrs[(size_t)e * 4 + 1];
    ev0S[wave][lane * 3 + 1] = edge_attrs[(size_t)e * 4 + 2];
    ev0S[wave][lane * 3 + 2] = edge_attrs[(size_t)e * 4 + 3];
    cutS[wave][lane] = cutoff_of(edge_lengths[e]);
  }
  DSWAIT();

  // lat_in tile: 16 rows x 16 dwords (K=32 with zero pad above K=16)
  for (int i = lane; i < 256; i += 32) {
    int m = i >> 4, t = i & 15;
    float a = 0.f, b = 0.f;
    if (t < 8) {
      int e = e0 + m;
      float vv[2];
#pragma unroll
      for (int q = 0; q < 2; ++q) {
        int j = 2 * t + q;
        float v;
        if (j < 4)      v = node_attrs[(size_t)ctrS[wave][m] * 4 + j];
        else if (j < 8) v = node_attrs[(size_t)nbrS[wave][m] * 4 + (j - 4)];
        else            v = edge_embed[(size_t)e * 8 + (j - 8)];
        vv[q] = v;
      }
      a = vv[0]; b = vv[1];
    }
    act[m * 64 + t] = pk2(a, b);
  }
  DSWAIT();

  v8f C[8];
  gemm<1, 8>(act, 64, wPack + OFF_WL0A, lane, C);
#pragma unroll
  for (int nt = 0; nt < 8; ++nt) {
#pragma unroll
    for (int r = 0; r < 8; ++r) C[nt][r] = silu(C[nt][r]);
    stActRow(acth, 128, lane, nt, C[nt]);
  }
  DSWAIT();
  gemm<4, 8>(act, 64, wPack + OFF_WL0B, lane, C);
#pragma unroll
  for (int nt = 0; nt < 8; ++nt) {
#pragma unroll
    for (int r = 0; r < 8; ++r) C[nt][r] = silu(C[nt][r]);
    stActRow(acth, 128, lane, nt, C[nt]);
  }
  DSWAIT();
  gemm<4, 8>(act, 64, wPack + OFF_WL0C, lane, C);
  // latent0 = mlp_out * cutoff -> fp32 to out[:,0:128] and bf16 to act
#pragma unroll
  for (int nt = 0; nt < 8; ++nt) {
#pragma unroll
    for (int r = 0; r < 8; ++r) {
      int m = 8 * h + r;
      float v = C[nt][r] * cutS[wave][m];
      C[nt][r] = v;
      out[(size_t)(e0 + m) * 256 + nt * 16 + l16] = v;
    }
    stActRow(acth, 128, lane, nt, C[nt]);
  }
  DSWAIT();
  gemm<4, 8>(act, 64, wPack + OFF_WENV0, lane, C);  // w = latent @ Wenv0
#pragma unroll
  for (int nt = 0; nt < 8; ++nt) {
#pragma unroll
    for (int r = 0; r < 8; ++r) {
      int m = 8 * h + r, e = e0 + m;
      float val = C[nt][r];
      float es = es0S[wave][m];
      if (nt < 2) {  // fs = es0 * w[:, 0:32]
        fsBuf[(size_t)e * 32 + nt * 16 + l16] = es * val;
      } else if (nt < 4) {  // fv = ev0 (x) w[:, 32:64]
        int k = (nt - 2) * 16 + l16;
        fvBuf[(size_t)e * 96 + k * 3 + 0] = ev0S[wave][m * 3 + 0] * val;
        fvBuf[(size_t)e * 96 + k * 3 + 1] = ev0S[wave][m * 3 + 1] * val;
        fvBuf[(size_t)e * 96 + k * 3 + 2] = ev0S[wave][m * 3 + 2] * val;
      } else if (nt < 6) {  // env_s accumulation: w[:, 64:96]
        int k = (nt - 4) * 16 + l16;
        atomicAdd(&envRaw[(size_t)ctrS[wave][m] * 128 + k], es * val);
      } else {  // env_v accumulation: w[:, 96:128]
        int k = (nt - 6) * 16 + l16;
        float* dst = &envRaw[(size_t)ctrS[wave][m] * 128 + 32 + k * 3];
        atomicAdd(dst + 0, ev0S[wave][m * 3 + 0] * val);
        atomicAdd(dst + 1, ev0S[wave][m * 3 + 1] * val);
        atomicAdd(dst + 2, ev0S[wave][m * 3 + 2] * val);
      }
    }
  }
}

// ---------------- node kernel: env linear (32x32) + re-zero accumulator ----------------
__global__ void k_node(const float* __restrict__ Ws, const float* __restrict__ Wv,
                       float* __restrict__ envRaw, float* __restrict__ envLin, int N) {
  int t = blockIdx.x * blockDim.x + threadIdx.x;
  if (t >= N * 32) return;
  int n = t >> 5, k = t & 31;  // one wave per node
  const float scale = (1.f / 16.f) * 0.17677669529663687f;  // inv_avg / sqrt(32)
  const float* raw = envRaw + (size_t)n * 128;
  float ss = 0.f, a0 = 0.f, a1 = 0.f, a2 = 0.f;
  for (int m = 0; m < 32; ++m) {
    float ws = Ws[m * 32 + k], wv = Wv[m * 32 + k];
    ss += raw[m] * ws;
    a0 += raw[32 + m * 3 + 0] * wv;
    a1 += raw[32 + m * 3 + 1] * wv;
    a2 += raw[32 + m * 3 + 2] * wv;
  }
  LDWAIT();  // all wave loads of raw complete before we zero it
  float* lin = envLin + (size_t)n * 128;
  lin[k] = ss * scale;
  lin[32 + k * 3 + 0] = a0 * scale;
  lin[32 + k * 3 + 1] = a1 * scale;
  lin[32 + k * 3 + 2] = a2 * scale;
  float* rw = envRaw + (size_t)n * 128;
  rw[k] = 0.f;
  rw[32 + k * 3 + 0] = 0.f; rw[32 + k * 3 + 1] = 0.f; rw[32 + k * 3 + 2] = 0.f;
}

// ---------------- edge kernel, layer 1 ----------------
__global__ __launch_bounds__(TPB) void k_edge1(
    const float* __restrict__ edge_attrs, const float* __restrict__ edge_lengths,
    const int* __restrict__ edge_index, const unsigned* __restrict__ wPack,
    float* __restrict__ out, float* __restrict__ fsBuf, float* __restrict__ fvBuf,
    const float* __restrict__ envLin, float* __restrict__ envRaw, int E) {
  int wave = threadIdx.x >> 5, lane = threadIdx.x & 31;
  int e0 = (blockIdx.x * WAVES + wave) * 16;
  if (e0 >= E) return;

  __shared__ unsigned actS[WAVES][16 * 96];  // K=192: latent0(0..63) | scalars(64..95)
  __shared__ unsigned ovS[WAVES][3 * 512];
  __shared__ int ctrS[WAVES][16];
  __shared__ float es0S[WAVES][16], ev0S[WAVES][48], cutS[WAVES][16];

  unsigned* act = actS[wave];
  unsigned* ov = ovS[wave];
  unsigned short* acth = (unsigned short*)act;
  int h = lane >> 4, l16 = lane & 15;

  if (lane < 16) {
    int e = e0 + lane;
    ctrS[wave][lane] = edge_index[e];
    es0S[wave][lane] = edge_attrs[(size_t)e * 4 + 0];
    ev0S[wave][lane * 3 + 0] = edge_attrs[(size_t)e * 4 + 1];
    ev0S[wave][lane * 3 + 1] = edge_attrs[(size_t)e * 4 + 2];
    ev0S[wave][lane * 3 + 2] = edge_attrs[(size_t)e * 4 + 3];
    cutS[wave][lane] = cutoff_of(edge_lengths[e]);
  }
  DSWAIT();

  // out_s -> act[64..95] (doubles as "scalars" input of MLP1), out_v -> ov
  build_out_tiles(lane, e0, ctrS[wave], fsBuf, fvBuf, envLin, act + 64, 96, ov);
  DSWAIT();

  v8f C2[2];
  gemm<2, 2>(act + 64, 96, wPack + OFF_WLS0, lane, C2);  // fs1
#pragma unroll
  for (int nt = 0; nt < 2; ++nt)
#pragma unroll
    for (int r = 0; r < 8; ++r)
      fsBuf[(size_t)(e0 + 8 * h + r) * 32 + nt * 16 + l16] = C2[nt][r];
#pragma unroll
  for (int v = 0; v < 3; ++v) {  // fv1
    gemm<2, 2>(ov + v * 512, 32, wPack + OFF_WLV0, lane, C2);
#pragma unroll
    for (int nt = 0; nt < 2; ++nt)
#pragma unroll
      for (int r = 0; r < 8; ++r)
        fvBuf[(size_t)(e0 + 8 * h + r) * 96 + (nt * 16 + l16) * 3 + v] = C2[nt][r];
  }

  // latent0 cols 0..63 (fp32 in out[:,0:64]) -> act[0..31] bf16
  for (int i = lane; i < 512; i += 32) {
    int m = i >> 5, t = i & 31;
    size_t base = (size_t)(e0 + m) * 256;
    act[m * 96 + t] = pk2(out[base + 2 * t], out[base + 2 * t + 1]);
  }
  DSWAIT();

  v8f C[8];
  gemm<6, 8>(act, 96, wPack + OFF_WL1A, lane, C);  // 192 -> 128
#pragma unroll
  for (int nt = 0; nt < 8; ++nt) {
#pragma unroll
    for (int r = 0; r < 8; ++r) C[nt][r] = silu(C[nt][r]);
    stActRow(acth, 192, lane, nt, C[nt]);
  }
  DSWAIT();
  gemm<4, 8>(act, 96, wPack + OFF_WL1B, lane, C);
#pragma unroll
  for (int nt = 0; nt < 8; ++nt) {
#pragma unroll
    for (int r = 0; r < 8; ++r) C[nt][r] = silu(C[nt][r]);
    stActRow(acth, 192, lane, nt, C[nt]);
  }
  DSWAIT();
  gemm<4, 8>(act, 96, wPack + OFF_WL1C, lane, C);
  // latent1 = (latent0 + new_lat*cutoff) / sqrt(2) -> out[:,0:128] + act bf16
  const float is2 = 0.70710678118654752f;
#pragma unroll
  for (int nt = 0; nt < 8; ++nt) {
#pragma unroll
    for (int r = 0; r < 8; ++r) {
      int m = 8 * h + r;
      size_t o = (size_t)(e0 + m) * 256 + nt * 16 + l16;
      float v = (out[o] + C[nt][r] * cutS[wave][m]) * is2;
      C[nt][r] = v;
      out[o] = v;
    }
    stActRow(acth, 192, lane, nt, C[nt]);
  }
  DSWAIT();
  v8f W4[4];
  gemm<4, 4>(act, 96, wPack + OFF_WENV1, lane, W4);  // w_env1 (64 cols)
#pragma unroll
  for (int nt = 0; nt < 4; ++nt) {
#pragma unroll
    for (int r = 0; r < 8; ++r) {
      int m = 8 * h + r;
      float val = W4[nt][r];
      if (nt < 2) {
        atomicAdd(&envRaw[(size_t)ctrS[wave][m] * 128 + nt * 16 + l16],
                  es0S[wave][m] * val);
      } else {
        int k = (nt - 2) * 16 + l16;
        float* dst = &envRaw[(size_t)ctrS[wave][m] * 128 + 32 + k * 3];
        atomicAdd(dst + 0, ev0S[wave][m * 3 + 0] * val);
        atomicAdd(dst + 1, ev0S[wave][m * 3 + 1] * val);
        atomicAdd(dst + 2, ev0S[wave][m * 3 + 2] * val);
      }
    }
  }
}

// ---------------- edge kernel, final ----------------
__global__ __launch_bounds__(TPB) void k_edge2(
    const int* __restrict__ edge_index, const unsigned* __restrict__ wPack,
    float* __restrict__ out, const float* __restrict__ fsBuf,
    const float* __restrict__ fvBuf, const float* __restrict__ envLin, int E) {
  int wave = threadIdx.x >> 5, lane = threadIdx.x & 31;
  int e0 = (blockIdx.x * WAVES + wave) * 16;
  if (e0 >= E) return;

  __shared__ unsigned osS[WAVES][512];
  __shared__ unsigned ovS[WAVES][3 * 512];
  __shared__ int ctrS[WAVES][16];
  unsigned* os = osS[wave];
  unsigned* ov = ovS[wave];
  int h = lane >> 4, l16 = lane & 15;

  if (lane < 16) ctrS[wave][lane] = edge_index[e0 + lane];
  DSWAIT();
  build_out_tiles(lane, e0, ctrS[wave], fsBuf, fvBuf, envLin, os, 32, ov);
  DSWAIT();

  v8f C2[2];
  gemm<2, 2>(os, 32, wPack + OFF_WLS1, lane, C2);  // fs final -> eq_out[:, :, 0]
#pragma unroll
  for (int nt = 0; nt < 2; ++nt)
#pragma unroll
    for (int r = 0; r < 8; ++r)
      out[(size_t)(e0 + 8 * h + r) * 256 + 128 + (nt * 16 + l16) * 4 + 0] = C2[nt][r];
#pragma unroll
  for (int v = 0; v < 3; ++v) {  // fv final -> eq_out[:, :, 1+v]
    gemm<2, 2>(ov + v * 512, 32, wPack + OFF_WLV1, lane, C2);
#pragma unroll
    for (int nt = 0; nt < 2; ++nt)
#pragma unroll
      for (int r = 0; r < 8; ++r)
        out[(size_t)(e0 + 8 * h + r) * 256 + 128 + (nt * 16 + l16) * 4 + 1 + v] = C2[nt][r];
  }
}

// ---------------- host ----------------
extern "C" void kernel_launch(void* const* d_in, const int* in_sizes, int n_in,
                              void* d_out, int out_size, void* d_ws, size_t ws_size,
                              hipStream_t stream) {
  const float* edge_attrs   = (const float*)d_in[0];
  const float* edge_embed   = (const float*)d_in[1];
  const float* node_attrs   = (const float*)d_in[2];
  const float* edge_lengths = (const float*)d_in[3];
  const int*   edge_index   = (const int*)d_in[4];
  const float* Wl0a = (const float*)d_in[5];
  const float* Wl0b = (const float*)d_in[6];
  const float* Wl0c = (const float*)d_in[7];
  const float* Wl1a = (const float*)d_in[8];
  const float* Wl1b = (const float*)d_in[9];
  const float* Wl1c = (const float*)d_in[10];
  const float* Wenv0 = (const float*)d_in[11];
  const float* Wenv1 = (const float*)d_in[12];
  const float* Wes0 = (const float*)d_in[13];
  const float* Wes1 = (const float*)d_in[14];
  const float* Wev0 = (const float*)d_in[15];
  const float* Wev1 = (const float*)d_in[16];
  const float* Wls0 = (const float*)d_in[17];
  const float* Wls1 = (const float*)d_in[18];
  const float* Wlv0 = (const float*)d_in[19];
  const float* Wlv1 = (const float*)d_in[20];

  int E = in_sizes[3];
  int N = in_sizes[2] / 4;
  float* out = (float*)d_out;
  unsigned* wPack = (unsigned*)d_ws;
  float* wsF = (float*)d_ws;

  float* fsBuf  = wsF + WS_FLOAT_BASE;
  float* fvBuf  = fsBuf + (size_t)E * 32;
  float* envRaw = fvBuf + (size_t)E * 96;
  float* envLin = envRaw + (size_t)N * 128;

  const int PT = 256;
  auto pack = [&](const float* src, int off, int Kp, int srcK, int Nc, float sc) {
    int KT = Kp / 32;
    int total = KT * (Nc / 16) * 256;
    k_pack<<<(total + PT - 1) / PT, PT, 0, stream>>>(src, wPack + off, KT, srcK, Nc, sc);
  };
  pack(Wl0a, OFF_WL0A, 32, 16, 128, 1.f / sqrtf(16.f));
  pack(Wl0b, OFF_WL0B, 128, 128, 128, 1.f / sqrtf(128.f));
  pack(Wl0c, OFF_WL0C, 128, 128, 128, 1.f / sqrtf(128.f));
  pack(Wl1a, OFF_WL1A, 192, 192, 128, 1.f / sqrtf(192.f));
  pack(Wl1b, OFF_WL1B, 128, 128, 128, 1.f / sqrtf(128.f));
  pack(Wl1c, OFF_WL1C, 128, 128, 128, 1.f / sqrtf(128.f));
  pack(Wenv0, OFF_WENV0, 128, 128, 128, 1.f / sqrtf(128.f));
  pack(Wenv1, OFF_WENV1, 128, 128, 64, 1.f / sqrtf(128.f));
  pack(Wls0, OFF_WLS0, 64, 64, 32, 1.f / sqrtf(64.f));
  pack(Wls1, OFF_WLS1, 64, 64, 32, 1.f / sqrtf(64.f));
  pack(Wlv0, OFF_WLV0, 64, 64, 32, 1.f / sqrtf(64.f));
  pack(Wlv1, OFF_WLV1, 64, 64, 32, 1.f / sqrtf(64.f));

  k_zero<<<(N * 128 + PT - 1) / PT, PT, 0, stream>>>(envRaw, N * 128);

  int tiles = E / 16;
  int blocks = (tiles + WAVES - 1) / WAVES;

  k_edge0<<<blocks, TPB, 0, stream>>>(edge_attrs, edge_embed, node_attrs, edge_lengths,
                                      edge_index, wPack, out, fsBuf, fvBuf, envRaw, E);
  k_node<<<(N * 32 + PT - 1) / PT, PT, 0, stream>>>(Wes0, Wev0, envRaw, envLin, N);
  k_edge1<<<blocks, TPB, 0, stream>>>(edge_attrs, edge_lengths, edge_index, wPack, out,
                                      fsBuf, fvBuf, envLin, envRaw, E);
  k_node<<<(N * 32 + PT - 1) / PT, PT, 0, stream>>>(Wes1, Wev1, envRaw, envLin, N);
  k_edge2<<<blocks, TPB, 0, stream>>>(edge_index, wPack, out, fsBuf, fvBuf, envLin, E);
}
